// DiscreteTernaryBlock_15058155339888
// MI455X (gfx1250) — matrix operand order repaired and checked by reading, not compiled, not run on hardware
//
#include <hip/hip_runtime.h>

// ---------------------------------------------------------------------------
// MI455X / gfx1250 implementation of the ternary transformer block.
// All matmuls use v_wmma_f32_16x16x32_bf16 (wave32). Weights (ternary f32,
// exact in bf16) are converted in-register; activations staged as bf16.
//
// Roofline: ~310 GFLOP vs 256 MB compulsory f32 weight traffic @ 23.3 TB/s
// -> weight-bandwidth bound. GEMM is tiled so each weight fragment is loaded
// once per block and amortized over 4 WMMAs (wave tile 64x16).
// ---------------------------------------------------------------------------

typedef __bf16 bf16;
typedef __attribute__((ext_vector_type(8)))  bf16  v8bf;
typedef __attribute__((ext_vector_type(16))) bf16  v16bf;
typedef __attribute__((ext_vector_type(8)))  float v8f;

static constexpr int S   = 2048;
static constexpr int D   = 2048;
static constexpr int H   = 16;
static constexpr int DH  = 128;
static constexpr int DFF = 8192;

__device__ __forceinline__ v16bf concat8(v8bf lo, v8bf hi) {
  return __builtin_shufflevector(lo, hi, 0,1,2,3,4,5,6,7,8,9,10,11,12,13,14,15);
}

// A-fragment (16x32 bf16, MxK). lane&15 selects M row (caller folds into `row`),
// lane>>4 selects K 0..15 vs 16..31.
__device__ __forceinline__ v16bf ldA_bf(const bf16* __restrict__ base, int row,
                                        int lda, int k0, int lane) {
  const int half = lane >> 4;
  const bf16* p = base + row * lda + k0 + half * 16;
  v8bf lo = *(const v8bf*)(p);
  v8bf hi = *(const v8bf*)(p + 8);
  return concat8(lo, hi);
}

// A-fragment built from f32 (LDS softmax probabilities).
__device__ __forceinline__ v16bf ldA_f32(const float* __restrict__ base, int row,
                                         int lda, int k0, int lane) {
  const int half = lane >> 4;
  const float* p = base + row * lda + k0 + half * 16;
  v16bf a;
#pragma unroll
  for (int i = 0; i < 8; ++i) {
    a[i]     = (bf16)p[i];
    a[i + 8] = (bf16)p[i + 8];
  }
  return a;
}

// B-fragment (32x16 bf16, KxN). lane&15 selects the N column, which is a row of
// the K-contiguous source (W row / K row / Vt row). lane>>4 selects K 0..15 vs 16..31.
__device__ __forceinline__ v16bf ldB_bf(const bf16* __restrict__ base, int row,
                                        int ldb, int k0, int lane) {
  const bf16* p = base + row * ldb + k0 + (lane >> 4) * 16;
  v8bf lo = *(const v8bf*)(p);
  v8bf hi = *(const v8bf*)(p + 8);
  return concat8(lo, hi);
}

// B-fragment from ternary f32 weights, converted in-register (exact for {-1,0,1}).
__device__ __forceinline__ v16bf ldB_f32(const float* __restrict__ base, int row,
                                         int ldb, int k0, int lane) {
  const float* p = base + row * ldb + k0 + (lane >> 4) * 16;
  v16bf b;
#pragma unroll
  for (int i = 0; i < 16; ++i) b[i] = (bf16)p[i];
  return b;
}

__device__ __forceinline__ v8f wmma_bf16(v16bf a, v16bf b, v8f c) {
  return __builtin_amdgcn_wmma_f32_16x16x32_bf16(false, a, false, b,
                                                 (short)0, c, false, false);
}

// ---------------------------------------------------------------------------
// RMSNorm: f32 row -> bf16 row.
// ---------------------------------------------------------------------------
__global__ void rmsnorm_k(const float* __restrict__ x, const float* __restrict__ w,
                          bf16* __restrict__ out, int d) {
  const int row = blockIdx.x;
  const float* xr = x + (size_t)row * d;
  __shared__ float red[256];
  float ss = 0.f;
  for (int i = threadIdx.x; i < d; i += 256) { float v = xr[i]; ss += v * v; }
  red[threadIdx.x] = ss;
  __syncthreads();
  for (int off = 128; off > 0; off >>= 1) {
    if (threadIdx.x < off) red[threadIdx.x] += red[threadIdx.x + off];
    __syncthreads();
  }
  const float inv = rsqrtf(red[0] / (float)d + 1e-6f);
  for (int i = threadIdx.x; i < d; i += 256)
    out[(size_t)row * d + i] = (bf16)(xr[i] * inv * w[i]);
}

// ---------------------------------------------------------------------------
// Generic ternary GEMM: C[M,N] = (A[M,K] @ W[N,K]^T) * s  (+ epilogue)
// Block = 256 threads = 8 waves. Block tile 64(M) x 128(N); each wave owns a
// distinct 16-column slab of W and computes a 64x16 strip (4 m-tiles), so every
// weight fragment is fetched once per block and feeds 4 WMMAs (the weight
// stream is the roofline-limiting traffic; A is cheap bf16).
// MODE 0: bf16 out, row-major ldo=N          (gate / up)
// MODE 1: bf16 out, head layout (h,s,dh)     (Q, K)
// MODE 2: bf16 out, transposed (h,dh,s)      (V)
// MODE 3: f32 out = resid + c                (O-proj -> x2, down-proj -> final)
// ---------------------------------------------------------------------------
template <int MODE>
__global__ void gemm_tern(const bf16* __restrict__ A, const float* __restrict__ W,
                          const float* __restrict__ scale_p, void* __restrict__ out,
                          const float* __restrict__ resid, int M, int N, int K) {
  const int lane  = threadIdx.x & 31;
  const int wv    = threadIdx.x >> 5;          // 0..7 -> n-slab within block
  const int mbase = blockIdx.y * 64;
  const int ncol  = blockIdx.x * 128 + wv * 16;

  const float s = *scale_p;

  const int brow = ncol + (lane & 15);         // W row = output column
  const int am   = mbase + (lane & 15);

  v8f acc[4] = {v8f{}, v8f{}, v8f{}, v8f{}};
  for (int k0 = 0; k0 < K; k0 += 32) {
    // keep the weight stream warm (global_prefetch_b8)
    __builtin_prefetch(W + (size_t)brow * K + k0 + 256, 0, 1);
    v16bf b = ldB_f32(W, brow, K, k0, lane);   // one weight fragment ...
#pragma unroll
    for (int mt = 0; mt < 4; ++mt) {           // ... amortized over 4 WMMAs
      v16bf a = ldA_bf(A, am + mt * 16, K, k0, lane);
      acc[mt] = wmma_bf16(a, b, acc[mt]);
    }
  }

  const int half = lane >> 4;
  const int nn   = lane & 15;
  const int col  = ncol + nn;
#pragma unroll
  for (int mt = 0; mt < 4; ++mt) {
#pragma unroll
    for (int r = 0; r < 8; ++r) {
      const int row = mbase + mt * 16 + r + half * 8;
      const float c = acc[mt][r] * s;
      if (MODE == 0) {
        ((bf16*)out)[(size_t)row * N + col] = (bf16)c;
      } else if (MODE == 1) {
        const int h = col >> 7, dh = col & (DH - 1);
        ((bf16*)out)[((size_t)h * S + row) * DH + dh] = (bf16)c;
      } else if (MODE == 2) {
        const int h = col >> 7, dh = col & (DH - 1);
        ((bf16*)out)[((size_t)h * DH + dh) * S + row] = (bf16)c;
      } else {
        ((float*)out)[(size_t)row * N + col] =
            resid[(size_t)row * N + col] + c;
      }
    }
  }
}

// ---------------------------------------------------------------------------
// Flash-style causal attention, one block per (16-query tile, head).
// Scores (16 x 2048, f32) live in dynamic LDS; two-pass softmax; WMMA for
// both QK^T and P·V. Epilogue adds alpha[h] * normed (x_heads residual).
// ---------------------------------------------------------------------------
__global__ void attn_k(const bf16* __restrict__ Q, const bf16* __restrict__ Kb,
                       const bf16* __restrict__ Vt, const bf16* __restrict__ normed,
                       const float* __restrict__ alpha, bf16* __restrict__ attnflat) {
  const int qt = blockIdx.x;       // 0..127
  const int h  = blockIdx.y;       // 0..15
  const int q0 = qt * 16;
  const int tid  = threadIdx.x;
  const int lane = tid & 31;
  const int wv   = tid >> 5;
  const int m    = lane & 15;

  extern __shared__ float probs[];     // [16][S] f32
  __shared__ float red[16][16];

  const bf16* Qh = Q  + ((size_t)h * S + q0) * DH;  // 16 x 128, lda = DH
  const bf16* Kh = Kb + (size_t)h * S * DH;         // [key][dh], lda = DH
  const bf16* Vh = Vt + (size_t)h * DH * S;         // [dh][key], lda = S

  const float sscale = 0.08838834764831845f;        // 1/sqrt(128)

  // ---- score phase: wave wv covers key tiles [wv*16, wv*16+16) ----
  // (compiler hoists the 4 Q fragments out of the kt loop)
  for (int kt = wv * 16; kt < wv * 16 + 16; ++kt) {
    const int kbase = kt * 16;
    v8f acc = {};
#pragma unroll
    for (int c = 0; c < DH / 32; ++c) {
      v16bf a = ldA_bf(Qh, m, DH, c * 32, lane);
      v16bf b = ldB_bf(Kh, kbase + m, DH, c * 32, lane);
      acc = wmma_bf16(a, b, acc);
    }
    const int half = lane >> 4, nn = lane & 15;
#pragma unroll
    for (int r = 0; r < 8; ++r) {
      const int qq = r + half * 8;
      probs[qq * S + kbase + nn] = acc[r] * sscale;
    }
  }
  __syncthreads();

  // ---- two-pass causal softmax: 16 threads per query row ----
  {
    const int r = tid >> 4, l16 = tid & 15;
    const int qg = q0 + r;
    float* row = probs + r * S;
    float mx = -3.4e38f;
    for (int c = l16; c < S; c += 16)
      if (c <= qg) mx = fmaxf(mx, row[c]);
    red[r][l16] = mx;
    __syncthreads();
    mx = red[r][0];
#pragma unroll
    for (int i = 1; i < 16; ++i) mx = fmaxf(mx, red[r][i]);
    __syncthreads();
    float sum = 0.f;
    for (int c = l16; c < S; c += 16) {
      float p = (c <= qg) ? __expf(row[c] - mx) : 0.f;
      row[c] = p;
      sum += p;
    }
    red[r][l16] = sum;
    __syncthreads();
    sum = 0.f;
#pragma unroll
    for (int i = 0; i < 16; ++i) sum += red[r][i];
    const float rinv = 1.f / sum;
    for (int c = l16; c < S; c += 16) row[c] *= rinv;
  }
  __syncthreads();

  // ---- P·V phase: wave wv computes dh tile [wv*16, wv*16+16) ----
  const int dhb = wv * 16;
  v8f acc = {};
  for (int k0 = 0; k0 < S; k0 += 32) {
    v16bf a = ldA_f32(probs, m, S, k0, lane);
    v16bf b = ldB_bf(Vh, dhb + m, S, k0, lane);
    acc = wmma_bf16(a, b, acc);
  }

  const float al = alpha[h];
  const int half = lane >> 4, nn = lane & 15;
#pragma unroll
  for (int r = 0; r < 8; ++r) {
    const int q   = q0 + r + half * 8;
    const int col = h * DH + dhb + nn;
    const float c = acc[r] + al * (float)normed[(size_t)q * D + col];
    attnflat[(size_t)q * D + col] = (bf16)c;
  }
}

// ---------------------------------------------------------------------------
// hidden = silu(g) * u   (bf16 in/out)
// ---------------------------------------------------------------------------
__global__ void silu_mul_k(const bf16* __restrict__ g, const bf16* __restrict__ u,
                           bf16* __restrict__ hdn, int n) {
  const int i = blockIdx.x * blockDim.x + threadIdx.x;
  if (i < n) {
    const float gv = (float)g[i];
    const float uv = (float)u[i];
    hdn[i] = (bf16)((gv / (1.f + __expf(-gv))) * uv);
  }
}

// ---------------------------------------------------------------------------
extern "C" void kernel_launch(void* const* d_in, const int* in_sizes, int n_in,
                              void* d_out, int out_size, void* d_ws, size_t ws_size,
                              hipStream_t stream) {
  const float* x     = (const float*)d_in[0];
  const float* naw   = (const float*)d_in[1];
  const float* nmw   = (const float*)d_in[2];
  const float* Wq    = (const float*)d_in[3];  const float* sq = (const float*)d_in[4];
  const float* Wk    = (const float*)d_in[5];  const float* sk = (const float*)d_in[6];
  const float* Wv    = (const float*)d_in[7];  const float* sv = (const float*)d_in[8];
  const float* Wo    = (const float*)d_in[9];  const float* so = (const float*)d_in[10];
  const float* Wg    = (const float*)d_in[11]; const float* sg = (const float*)d_in[12];
  const float* Wu    = (const float*)d_in[13]; const float* su = (const float*)d_in[14];
  const float* Wd    = (const float*)d_in[15]; const float* sd = (const float*)d_in[16];
  const float* alpha = (const float*)d_in[17];
  float* out = (float*)d_out;

  char* ws = (char*)d_ws;
  const size_t MB = 1u << 20;
  float* x2      = (float*)(ws);                 // 16 MB, live whole call
  bf16* normed   = (bf16*)(ws + 16 * MB);        //  8 MB (normed1, then normed2)
  // phase 1 region (24..64 MB): QKV + attnflat
  bf16* qb       = (bf16*)(ws + 24 * MB);        //  8 MB
  bf16* kb       = (bf16*)(ws + 32 * MB);        //  8 MB
  bf16* vt       = (bf16*)(ws + 40 * MB);        //  8 MB (transposed h,dh,s)
  bf16* attnflat = (bf16*)(ws + 56 * MB);        //  8 MB
  // phase 2 reuse: G/U over the dead QKV region, hidden over dead attnflat
  bf16* gbuf     = (bf16*)(ws + 24 * MB);        // 32 MB
  bf16* ubuf     = (bf16*)(ws + 40 * MB);        // 32 MB
  bf16* hidden   = (bf16*)(ws + 56 * MB);        // 32 MB   (peak ws = 88 MB)

  const dim3 blk(256);

  // 1. normed1 = rmsnorm(x)
  rmsnorm_k<<<dim3(S), blk, 0, stream>>>(x, naw, normed, D);

  // 2-4. Q/K/V projections (head layouts)
  gemm_tern<1><<<dim3(D / 128, S / 64), blk, 0, stream>>>(normed, Wq, sq, qb, nullptr, S, D, D);
  gemm_tern<1><<<dim3(D / 128, S / 64), blk, 0, stream>>>(normed, Wk, sk, kb, nullptr, S, D, D);
  gemm_tern<2><<<dim3(D / 128, S / 64), blk, 0, stream>>>(normed, Wv, sv, vt, nullptr, S, D, D);

  // 5. causal attention + alpha * x_heads residual
  attn_k<<<dim3(S / 16, H), blk, 16 * S * sizeof(float), stream>>>(
      qb, kb, vt, normed, alpha, attnflat);

  // 6. x2 = x + attnflat @ Wo^T * so
  gemm_tern<3><<<dim3(D / 128, S / 64), blk, 0, stream>>>(attnflat, Wo, so, x2, x, S, D, D);

  // 7. normed2 = rmsnorm(x2)
  rmsnorm_k<<<dim3(S), blk, 0, stream>>>(x2, nmw, normed, D);

  // 8-9. gate / up projections
  gemm_tern<0><<<dim3(DFF / 128, S / 64), blk, 0, stream>>>(normed, Wg, sg, gbuf, nullptr, S, DFF, D);
  gemm_tern<0><<<dim3(DFF / 128, S / 64), blk, 0, stream>>>(normed, Wu, su, ubuf, nullptr, S, DFF, D);

  // 10. hidden = silu(g) * u
  silu_mul_k<<<dim3((S * DFF) / 256), blk, 0, stream>>>(gbuf, ubuf, hidden, S * DFF);

  // 11. out = x2 + hidden @ Wd^T * sd
  gemm_tern<3><<<dim3(D / 128, S / 64), blk, 0, stream>>>(hidden, Wd, sd, out, x2, S, D, DFF);
}
